// Decoder_59923383714463
// MI455X (gfx1250) — matrix operand order repaired and checked by reading, not compiled, not run on hardware
//
#include <hip/hip_runtime.h>
#include <hip/hip_bf16.h>

// ---------------------------------------------------------------------------
// Transformer decoder (B=4, C=512, M=512, H=8, K=V=64, F=2048, L=6, T=32000)
// All GEMMs via generic bf16 WMMA kernel (v_wmma_f32_16x16x32_bf16, fp32 acc).
// Tile: 128 x TN x 32, TN=128 (8 WMMA/wave/step) or TN=64 for N=64 GEMMs.
// B-access mode (row-major vs transposed) is a template parameter so each
// instantiation has straight-line staging. Staging: clause-grouped
// global_load_b128, v_perm_b32 bf16 packing, ds_store_b64; fragments via
// ds_load_b128.
// ---------------------------------------------------------------------------

#define B_ 4
#define C_ 512
#define M_ 512
#define H_ 8
#define KH 64
#define VH 64
#define F_ 2048
#define L_ 6
#define T_ 32000
#define BC (B_ * C_)

#define TM 128
#define TK 32
#define SA_S 40   // sA row stride (u16), 80B (multiple of 16B)

typedef __attribute__((ext_vector_type(16))) __bf16        v16bf;
typedef __attribute__((ext_vector_type(16))) unsigned short v16u;
typedef __attribute__((ext_vector_type(8)))  float          v8f;

// round-half-up bf16 conversion; two elements packed with one v_perm_b32
__device__ __forceinline__ unsigned short f2bf(float f) {
    unsigned u = __builtin_bit_cast(unsigned, f) + 0x8000u;
    return (unsigned short)(u >> 16);
}
__device__ __forceinline__ unsigned pk2bf(float a, float b) {
    unsigned ua = __builtin_bit_cast(unsigned, a) + 0x8000u;
    unsigned ub = __builtin_bit_cast(unsigned, b) + 0x8000u;
    // result bytes {ub[3],ub[2],ua[3],ua[2]} = (bf16(b)<<16) | bf16(a)
    return __builtin_amdgcn_perm(ub, ua, 0x07060302u);
}

// ---------------------------------------------------------------------------
// Generic batched GEMM, C[z] = act(A[z] x B[z] + bias)
//   A: a_off + row*lda + k                        (row-major fp32)
//   B: b_off + k*ldb + (n&63)*b_ns + (n>>6)*b_nh  (fp32; ldb==1 when BT)
//   C: c_off + row*ldc + n
//   z = blockIdx.z ; b = z/zH, h = z%zH
// ---------------------------------------------------------------------------
struct GemmP {
    const float* A; const float* B; float* C; const float* bias;
    int M, N, K;
    int lda; long long a_b, a_h;
    int ldb; int b_ns; long long b_nh, b_b, b_h;
    int ldc; long long c_b, c_h;
    int zH; int relu;
};

template <int TNN, bool BT>
__global__ __launch_bounds__(256) void gemm_bf16_wmma(GemmP p) {
    constexpr int NF  = TNN / 32;          // B frags per wave (2 or 4)
    constexpr int SBS = TNN + 8;           // sB row stride (u16), 16B multiple
    __shared__ unsigned short sA[TM][SA_S];   // [m][k] bf16 bits
    __shared__ unsigned short sB[TK][SBS];    // [k][n] bf16 bits

    const int tid  = threadIdx.x;
    const int z    = blockIdx.z;
    const int bi   = z / p.zH;
    const int hi   = z % p.zH;
    const long long a_off = (long long)bi * p.a_b + (long long)hi * p.a_h;
    const long long b_off = (long long)bi * p.b_b + (long long)hi * p.b_h;
    const long long c_off = (long long)bi * p.c_b + (long long)hi * p.c_h;

    const int m0 = blockIdx.y * TM;
    const int n0 = blockIdx.x * TNN;
    const int w    = tid >> 5;
    const int lane = tid & 31;
    const int wr = (w & 3) * 32;            // wave rows: 4 groups of 32
    const int wc = (w >> 2) * (TNN / 2);    // wave cols: 2 groups of TNN/2

    v8f acc[2][NF];
#pragma unroll
    for (int mi = 0; mi < 2; ++mi)
#pragma unroll
        for (int nj = 0; nj < NF; ++nj) acc[mi][nj] = (v8f){};

    const int am = lane & 15;
    const int kg = lane >> 4;

    // A staging: each thread loads 4x float4; row ar+32e, cols ac..ac+3
    const int ar = tid >> 3;
    const int ac = (tid & 7) * 4;
    const float* aPtr = p.A + a_off + (long long)(m0 + ar) * p.lda + ac;

    // B staging thread mapping
    const int bn_r  = (tid & (TNN / 4 - 1)) * 4;   // row-major path
    const int bk0_r = tid / (TNN / 4);
    const int bk4_t = (tid & 7) * 4;               // transposed path
    const int bn0_t = tid >> 3;

    // hoisted per-thread B base pointers
    const float* bPtrR = nullptr;
    const float* bPtrT[NF];
    if constexpr (!BT) {
        const long long Nl = n0 + bn_r;
        bPtrR = p.B + b_off + (Nl & 63) + (Nl >> 6) * p.b_nh;
    } else {
#pragma unroll
        for (int e = 0; e < NF; ++e) {
            long long Nl = n0 + bn0_t + 32 * e;
            bPtrT[e] = p.B + b_off + (Nl & 63) * (long long)p.b_ns
                     + (Nl >> 6) * p.b_nh + bk4_t;   // ldb == 1
        }
    }

    for (int k0 = 0; k0 < p.K; k0 += TK) {
        // ---- phase 1: clause-grouped global loads of A and B tiles --------
        float4 va[4];
#pragma unroll
        for (int e = 0; e < 4; ++e)
            va[e] = *(const float4*)(aPtr + (long long)(32 * e) * p.lda + k0);

        float4 vb[NF];
        if constexpr (!BT) {
            // row-major B: vectorize along n; 4-runs never cross a 64-col head
#pragma unroll
            for (int e = 0; e < NF; ++e) {
                int kk = bk0_r + e * (1024 / TNN);
                vb[e] = *(const float4*)(bPtrR + (long long)(k0 + kk) * p.ldb);
            }
        } else {
            // transposed B (ldb == 1): vectorize along k
#pragma unroll
            for (int e = 0; e < NF; ++e)
                vb[e] = *(const float4*)(bPtrT[e] + k0);
        }

        // ---- phase 2: convert to bf16 and store to LDS --------------------
        __syncthreads();
#pragma unroll
        for (int e = 0; e < 4; ++e) {
            uint2 pk; pk.x = pk2bf(va[e].x, va[e].y); pk.y = pk2bf(va[e].z, va[e].w);
            *(uint2*)&sA[ar + 32 * e][ac] = pk;
        }
        if constexpr (!BT) {
#pragma unroll
            for (int e = 0; e < NF; ++e) {
                int kk = bk0_r + e * (1024 / TNN);
                uint2 pk; pk.x = pk2bf(vb[e].x, vb[e].y); pk.y = pk2bf(vb[e].z, vb[e].w);
                *(uint2*)&sB[kk][bn_r] = pk;
            }
        } else {
#pragma unroll
            for (int e = 0; e < NF; ++e) {
                int n = bn0_t + 32 * e;
                sB[bk4_t + 0][n] = f2bf(vb[e].x);
                sB[bk4_t + 1][n] = f2bf(vb[e].y);
                sB[bk4_t + 2][n] = f2bf(vb[e].z);
                sB[bk4_t + 3][n] = f2bf(vb[e].w);
            }
        }
        __syncthreads();

        // A fragments: lane -> M row, lane>>4 -> K group; 2x ds_load_b128 each
        v16bf a[2];
#pragma unroll
        for (int mi = 0; mi < 2; ++mi) {
            v16u av;
            const int row = wr + mi * 16 + am;
#pragma unroll
            for (int e = 0; e < 16; ++e) {
                int kk = (e < 8) ? (kg * 8 + e) : (8 + kg * 8 + e);
                av[e] = sA[row][kk];
            }
            a[mi] = __builtin_bit_cast(v16bf, av);
        }
        // B fragments: lane -> N col, lane>>4 -> K group (same K order as A)
        v16bf b[NF];
#pragma unroll
        for (int nj = 0; nj < NF; ++nj) {
            v16u bv;
#pragma unroll
            for (int e = 0; e < 16; ++e) bv[e] = sB[(e < 8) ? (kg * 8 + e) : (8 + kg * 8 + e)][wc + nj * 16 + am];
            b[nj] = __builtin_bit_cast(v16bf, bv);
        }
#pragma unroll
        for (int mi = 0; mi < 2; ++mi)
#pragma unroll
            for (int nj = 0; nj < NF; ++nj)
                acc[mi][nj] = __builtin_amdgcn_wmma_f32_16x16x32_bf16(
                    false, a[mi], false, b[nj], (short)0, acc[mi][nj], false, false);
    }

    // C/D layout: lane&15 = N col, VGPR i -> M = i + 8*(lane>>4)
    const int cn  = lane & 15;
    const int crb = 8 * (lane >> 4);
#pragma unroll
    for (int mi = 0; mi < 2; ++mi) {
#pragma unroll
        for (int nj = 0; nj < NF; ++nj) {
            v8f acv = acc[mi][nj];
            int gn = n0 + wc + nj * 16 + cn;
            float bb = p.bias ? p.bias[gn] : 0.0f;
#pragma unroll
            for (int i = 0; i < 8; ++i) {
                int gr = m0 + wr + mi * 16 + crb + i;
                float v = acv[i] + bb;
                if (p.relu) v = fmaxf(v, 0.f);
                p.C[c_off + (long long)gr * p.ldc + gn] = v;
            }
        }
    }
}

// ---------------------------------------------------------------------------
// x[b,c,:] = embed[dec_in[b,c],:] + positional(c,:)
// ---------------------------------------------------------------------------
__global__ __launch_bounds__(256) void embed_pos_kernel(
    const int* __restrict__ dec_in, const float* __restrict__ embed,
    float* __restrict__ X) {
    int bc = blockIdx.x;
    int c  = bc % C_;
    int tok = dec_in[bc];
    for (int m = threadIdx.x; m < M_; m += 256) {
        float t   = (float)m / (float)(M_ - 1);
        float den = powf(10000.0f, t);
        float arg = (float)c / den;
        float pos = ((m & 1) == 0) ? sinf(arg) : cosf(arg);
        X[(long long)bc * M_ + m] = embed[(long long)tok * M_ + m] + pos;
    }
}

// ---------------------------------------------------------------------------
// Attention softmax over key axis (contiguous): S[row][c], row = (b*H+h)*C + d
// masked: add -100 where key c <= query d, scale by K^-0.5 before softmax.
// ---------------------------------------------------------------------------
__global__ __launch_bounds__(256) void attn_softmax_kernel(
    float* __restrict__ S, int masked, float scale) {
    long long row = blockIdx.x;
    int d = (int)(row & (C_ - 1));
    float* p = S + row * C_;
    int tid = threadIdx.x;
    float x0 = p[tid], x1 = p[tid + 256];
    if (masked) {
        if (tid       <= d) x0 += -100.0f;
        if (tid + 256 <= d) x1 += -100.0f;
    }
    x0 *= scale; x1 *= scale;
    __shared__ float red[256];
    red[tid] = fmaxf(x0, x1); __syncthreads();
    for (int s = 128; s > 0; s >>= 1) {
        if (tid < s) red[tid] = fmaxf(red[tid], red[tid + s]);
        __syncthreads();
    }
    float mx = red[0]; __syncthreads();
    float e0 = __expf(x0 - mx), e1 = __expf(x1 - mx);
    red[tid] = e0 + e1; __syncthreads();
    for (int s = 128; s > 0; s >>= 1) {
        if (tid < s) red[tid] += red[tid + s];
        __syncthreads();
    }
    float inv = 1.0f / red[0];
    p[tid] = e0 * inv; p[tid + 256] = e1 * inv;
}

// ---------------------------------------------------------------------------
// X = LayerNorm(X + A) * g + b  over last dim (512), one block per row
// ---------------------------------------------------------------------------
__global__ __launch_bounds__(256) void ln_residual_kernel(
    float* __restrict__ X, const float* __restrict__ A,
    const float* __restrict__ g, const float* __restrict__ bt) {
    long long row = blockIdx.x;
    float* xp = X + row * M_;
    const float* ap = A + row * M_;
    int tid = threadIdx.x;
    float y0 = xp[tid] + ap[tid];
    float y1 = xp[tid + 256] + ap[tid + 256];
    __shared__ float red[256];
    red[tid] = y0 + y1; __syncthreads();
    for (int s = 128; s > 0; s >>= 1) {
        if (tid < s) red[tid] += red[tid + s];
        __syncthreads();
    }
    float mu = red[0] * (1.0f / (float)M_); __syncthreads();
    float d0 = y0 - mu, d1 = y1 - mu;
    red[tid] = d0 * d0 + d1 * d1; __syncthreads();
    for (int s = 128; s > 0; s >>= 1) {
        if (tid < s) red[tid] += red[tid + s];
        __syncthreads();
    }
    float rstd = rsqrtf(red[0] * (1.0f / (float)M_) + 1e-5f);
    xp[tid]       = d0 * rstd * g[tid]       + bt[tid];
    xp[tid + 256] = d1 * rstd * g[tid + 256] + bt[tid + 256];
}

// ---------------------------------------------------------------------------
// In-place softmax over rows of length N (= 32000), online max/sum merge
// ---------------------------------------------------------------------------
__global__ __launch_bounds__(256) void softmax_rows_kernel(float* __restrict__ out, int N) {
    long long row = blockIdx.x;
    float* p = out + row * (long long)N;
    int tid = threadIdx.x;
    float m = -1e30f, s = 0.0f;
    for (int i = tid; i < N; i += 256) {
        float x = p[i];
        float nm = fmaxf(m, x);
        s = s * __expf(m - nm) + __expf(x - nm);
        m = nm;
    }
    __shared__ float sm[256], ss[256];
    sm[tid] = m; ss[tid] = s; __syncthreads();
    for (int st = 128; st > 0; st >>= 1) {
        if (tid < st) {
            float m2 = sm[tid + st], s2 = ss[tid + st];
            float nm = fmaxf(sm[tid], m2);
            ss[tid] = ss[tid] * __expf(sm[tid] - nm) + s2 * __expf(m2 - nm);
            sm[tid] = nm;
        }
        __syncthreads();
    }
    float mx = sm[0], inv = 1.0f / ss[0];
    for (int i = tid; i < N; i += 256) p[i] = __expf(p[i] - mx) * inv;
}

// ---------------------------------------------------------------------------
// Host-side launch helpers
// ---------------------------------------------------------------------------
static void launch_gemm(hipStream_t st,
                        const float* A, const float* B, float* C, const float* bias,
                        int M, int N, int K,
                        int lda, long long a_b, long long a_h,
                        int ldb, int b_ns, long long b_nh,
                        long long b_b, long long b_h,
                        int ldc, long long c_b, long long c_h,
                        int Z, int zH, int relu) {
    GemmP p{A, B, C, bias, M, N, K, lda, a_b, a_h,
            ldb, b_ns, b_nh, b_b, b_h, ldc, c_b, c_h, zH, relu};
    if (N % 128 == 0) {
        dim3 grid(N / 128, M / TM, Z);
        if (b_ns == 1)
            gemm_bf16_wmma<128, false><<<grid, dim3(256), 0, st>>>(p);
        else
            gemm_bf16_wmma<128, true><<<grid, dim3(256), 0, st>>>(p);
    } else {
        dim3 grid(N / 64, M / TM, Z);
        gemm_bf16_wmma<64, false><<<grid, dim3(256), 0, st>>>(p);  // PRE only
    }
}

// projection: out(BC x 512) = act(BC x 512) x W[h][m][k] (packed heads)
static void launch_proj(hipStream_t st, const float* act, const float* W, float* out) {
    launch_gemm(st, act, W, out, nullptr,
                BC, M_, M_,
                /*lda*/ M_, 0, 0,
                /*ldb*/ KH, /*b_ns*/ 1, /*b_nh*/ (long long)M_ * KH, 0, 0,
                /*ldc*/ M_, 0, 0, 1, 1, 0);
}

extern "C" void kernel_launch(void* const* d_in, const int* in_sizes, int n_in,
                              void* d_out, int out_size, void* d_ws, size_t ws_size,
                              hipStream_t stream) {
    const float* enc   = (const float*)d_in[0];
    const int*   decIn = (const int*)  d_in[1];
    const float* emb   = (const float*)d_in[2];
    const float* wq    = (const float*)d_in[3];
    const float* wk    = (const float*)d_in[4];
    const float* wv    = (const float*)d_in[5];
    const float* wo    = (const float*)d_in[6];
    const float* w1    = (const float*)d_in[7];
    const float* b1    = (const float*)d_in[8];
    const float* w2    = (const float*)d_in[9];
    const float* b2    = (const float*)d_in[10];
    const float* lng   = (const float*)d_in[11];
    const float* lnb   = (const float*)d_in[12];
    const float* wf    = (const float*)d_in[13];
    const float* bf    = (const float*)d_in[14];
    float* out = (float*)d_out;

    // workspace layout (floats)
    float* ws  = (float*)d_ws;
    float* X   = ws;                        // BC x 512
    float* Q   = X   + (size_t)BC * M_;     // BC x 512  (b,c | h,k)
    float* Kb  = Q   + (size_t)BC * M_;
    float* Vb  = Kb  + (size_t)BC * M_;
    float* PRE = Vb  + (size_t)BC * M_;     // BC x 512  (b,d | h,v)
    float* AO  = PRE + (size_t)BC * M_;
    float* FF1 = AO  + (size_t)BC * M_;     // BC x 2048
    float* FF2 = FF1 + (size_t)BC * F_;
    float* S   = FF2 + (size_t)BC * M_;     // B*H*C x C  (scores, stored as S^T)

    const long long HCC = (long long)H_ * C_ * C_;    // per-batch score stride
    const long long CC  = (long long)C_ * C_;         // per-head score stride
    const long long BCM = (long long)C_ * M_;         // per-batch act stride
    const float scale = 0.125f;                       // K^-0.5

    embed_pos_kernel<<<dim3(BC), dim3(256), 0, stream>>>(decIn, emb, X);

    for (int l = 0; l < L_; ++l) {
        const size_t attW = (size_t)H_ * M_ * KH;     // 262144 per (l,s)
        for (int s = 0; s < 2; ++s) {
            const float* Wq = wq + ((size_t)l * 2 + s) * attW;
            const float* Wk = wk + ((size_t)l * 2 + s) * attW;
            const float* Wv = wv + ((size_t)l * 2 + s) * attW;
            const float* Wo = wo + ((size_t)l * 2 + s) * attW;
            const float* kvsrc = (s == 0) ? X : enc;

            launch_proj(stream, X,     Wq, Q);
            launch_proj(stream, kvsrc, Wk, Kb);
            launch_proj(stream, kvsrc, Wv, Vb);

            // S^T[b,h,d,c] = sum_k Q[b,d,h,k] * K[b,c,h,k]
            launch_gemm(stream, Q, Kb, S, nullptr,
                        C_, C_, KH,
                        /*lda*/ M_, /*a_b*/ BCM, /*a_h*/ 64,
                        /*ldb*/ 1, /*b_ns*/ M_, /*b_nh*/ (long long)64 * M_,
                        /*b_b*/ BCM, /*b_h*/ 64,
                        /*ldc*/ C_, /*c_b*/ HCC, /*c_h*/ CC,
                        /*Z*/ B_ * H_, /*zH*/ H_, 0);

            attn_softmax_kernel<<<dim3(B_ * H_ * C_), dim3(256), 0, stream>>>(
                S, (s == 0) ? 1 : 0, scale);

            // PRE[b,d,h,v] = sum_c att[b,h,d,c] * V[b,c,h,v]
            launch_gemm(stream, S, Vb, PRE, nullptr,
                        C_, VH, C_,
                        /*lda*/ C_, /*a_b*/ HCC, /*a_h*/ CC,
                        /*ldb*/ M_, /*b_ns*/ 1, /*b_nh*/ 64,
                        /*b_b*/ BCM, /*b_h*/ 64,
                        /*ldc*/ M_, /*c_b*/ BCM, /*c_h*/ 64,
                        /*Z*/ B_ * H_, /*zH*/ H_, 0);

            // AO = PRE x Wo[h][v][m]
            launch_gemm(stream, PRE, Wo, AO, nullptr,
                        BC, M_, M_,
                        /*lda*/ M_, 0, 0,
                        /*ldb*/ M_, /*b_ns*/ 1, /*b_nh*/ 64,
                        0, 0,
                        /*ldc*/ M_, 0, 0, 1, 1, 0);

            ln_residual_kernel<<<dim3(BC), dim3(256), 0, stream>>>(
                X, AO, lng + ((size_t)l * 3 + s) * M_, lnb + ((size_t)l * 3 + s) * M_);
        }

        // FFN
        const float* W1 = w1 + (size_t)l * M_ * F_;
        const float* W2 = w2 + (size_t)l * F_ * M_;
        launch_gemm(stream, X, W1, FF1, b1 + (size_t)l * F_,
                    BC, F_, M_,
                    /*lda*/ M_, 0, 0,
                    /*ldb*/ F_, /*b_ns*/ 1, /*b_nh*/ 64,
                    0, 0,
                    /*ldc*/ F_, 0, 0, 1, 1, /*relu*/ 1);
        launch_gemm(stream, FF1, W2, FF2, b2 + (size_t)l * M_,
                    BC, M_, F_,
                    /*lda*/ F_, 0, 0,
                    /*ldb*/ M_, /*b_ns*/ 1, /*b_nh*/ 64,
                    0, 0,
                    /*ldc*/ M_, 0, 0, 1, 1, 0);
        ln_residual_kernel<<<dim3(BC), dim3(256), 0, stream>>>(
            X, FF2, lng + ((size_t)l * 3 + 2) * M_, lnb + ((size_t)l * 3 + 2) * M_);
    }

    // logits[b,c,t] = X x wf^T + bf   (wf is T x M row-major -> transposed access)
    launch_gemm(stream, X, wf, out, bf,
                BC, T_, M_,
                /*lda*/ M_, 0, 0,
                /*ldb*/ 1, /*b_ns*/ M_, /*b_nh*/ (long long)64 * M_,
                0, 0,
                /*ldc*/ T_, 0, 0, 1, 1, 0);

    softmax_rows_kernel<<<dim3(BC), dim3(256), 0, stream>>>(out, T_);
}
